// BGFG_OnOffEncoding_70153995813580
// MI455X (gfx1250) — matrix operand order repaired and checked by reading, not compiled, not run on hardware
//
#include <hip/hip_runtime.h>
#include <hip/hip_bf16.h>
#include <math.h>

// ---------------------------------------------------------------------------
// NeRF-style MLP, fully fused, driven by v_wmma_f32_16x16x32_f16.
// One wave (32 lanes) owns a 16-row tile; all activations stay in LDS.
// Weights are pre-packed (separate tiny kernel) into WMMA B-fragment order.
// Round 2: + global_prefetch of next tile's inputs, + hoisted views/keep
// loads so HBM latency overlaps the 38-WMMA compute chain.
// ---------------------------------------------------------------------------

typedef __attribute__((ext_vector_type(16))) _Float16 v16h;
typedef __attribute__((ext_vector_type(8)))  _Float16 v8h;
typedef __attribute__((ext_vector_type(8)))  float    v8f;

#define WMMA_F16(a, b, c) \
  __builtin_amdgcn_wmma_f32_16x16x32_f16(false, (a), false, (b), (short)0, (c), false, false)

// Packed-weight offsets (in halves) inside d_ws. Fragment = 512 halves (32Kx16N).
#define OFF_S0 0
#define OFF_S1 2048
#define OFF_S2 6144
#define OFF_C0 8192
#define OFF_C1 10240
#define OFF_C2 14336
#define OFF_C3 18432
#define PW_TOTAL 19456   // halves = 38912 bytes = 2432 uint4

#define NWAVES 4
#define TPW    8         // 16-row tiles per wave -> 512 rows per block

// DS ops from one wave are processed in order; this wait (plus the memory
// clobber) makes the cross-lane LDS handoff between layers explicit.
__device__ __forceinline__ void lds_fence() {
  asm volatile("s_wait_dscnt 0" ::: "memory");
}

// ---- weight pre-pack: f32 row-major (Kreal x Nreal) -> f16 B-fragments -----
// B fragment layout (16-bit B, 32x16): half h of lane l: hi=l/16
//   K_local = h<8 ? hi*8 + h : 16 + hi*8 + (h-8) ; N_local = l%16 (matches loadA)
__global__ void pack_weights(const float* __restrict__ w, _Float16* __restrict__ dst,
                             int Kreal, int Nreal, int Ktiles, int Ntiles) {
  int total = Ktiles * Ntiles * 512;
  for (int e = blockIdx.x * blockDim.x + threadIdx.x; e < total;
       e += gridDim.x * blockDim.x) {
    int frag = e >> 9, r = e & 511;
    int lane = r >> 4, h = r & 15;
    int hi = lane >> 4, nloc = lane & 15;
    int v = h >> 1, p = h & 1;
    int kloc = (h < 8) ? (hi * 8 + 2 * v + p) : (16 + hi * 8 + 2 * v + p - 8);
    int kt = frag / Ntiles, nt = frag - kt * Ntiles;
    int K = kt * 32 + kloc, Ncol = nt * 16 + nloc;
    float val = (K < Kreal && Ncol < Nreal) ? w[K * Nreal + Ncol] : 0.0f;
    dst[e] = (_Float16)val;
  }
}

// ---- fragment loaders ------------------------------------------------------
// A (16x32 f16, M=lane%16): two contiguous 8-half runs per lane:
//   halves 0..7  -> K = hi*8 + 0..7 ;  halves 8..15 -> K = 16 + hi*8 + 0..7
__device__ __forceinline__ v16h loadA(const _Float16* s, int C, int kt, int lane) {
  const int row = lane & 15, hi8 = (lane >> 4) << 3;
  const _Float16* base = s + row * C + kt * 32;
  v8h lo = *(const v8h*)(base + hi8);
  v8h hi = *(const v8h*)(base + 16 + hi8);
  return __builtin_shufflevector(lo, hi, 0,1,2,3,4,5,6,7,8,9,10,11,12,13,14,15);
}

// B fragments are stored pre-laned: lane*16 halves contiguous.
__device__ __forceinline__ v16h loadB(const _Float16* w, int frag, int lane) {
  return *(const v16h*)(w + frag * 512 + lane * 16);
}

// ---- one f16-in / f16-out layer, N=64 out (4 n-tiles), optional ReLU -------
// D layout: VGPR r holds (M=r, N=lane) lanes 0-15 / (M=r+8, N=lane-16) 16-31.
template <int KT, bool RELU>
__device__ __forceinline__ void layer_h64(const _Float16* __restrict__ w,
                                          const _Float16* __restrict__ in, int Cin,
                                          _Float16* __restrict__ out, int lane) {
  v16h a[KT];
#pragma unroll
  for (int kt = 0; kt < KT; ++kt) a[kt] = loadA(in, Cin, kt, lane);
  const int col = lane & 15;
  const int rbase = (lane >> 4) << 3;
#pragma unroll
  for (int nt = 0; nt < 4; ++nt) {
    v8f acc = {};
#pragma unroll
    for (int kt = 0; kt < KT; ++kt)
      acc = WMMA_F16(a[kt], loadB(w, kt * 4 + nt, lane), acc);
#pragma unroll
    for (int r = 0; r < 8; ++r) {
      float xv = acc[r];
      if (RELU) xv = fmaxf(xv, 0.0f);
      out[(rbase + r) * 64 + nt * 16 + col] = (_Float16)xv;
    }
  }
}

// sigma layer: 64 -> 17 (padded to 32), f32 out, no activation
__device__ __forceinline__ void layer_s2(const _Float16* __restrict__ w,
                                         const _Float16* __restrict__ in,
                                         float* __restrict__ outf, int lane) {
  v16h a0 = loadA(in, 64, 0, lane), a1 = loadA(in, 64, 1, lane);
  const int col = lane & 15;
  const int rbase = (lane >> 4) << 3;
#pragma unroll
  for (int nt = 0; nt < 2; ++nt) {
    v8f acc = {};
    acc = WMMA_F16(a0, loadB(w, 0 * 2 + nt, lane), acc);
    acc = WMMA_F16(a1, loadB(w, 1 * 2 + nt, lane), acc);
#pragma unroll
    for (int r = 0; r < 8; ++r) outf[(rbase + r) * 32 + nt * 16 + col] = acc[r];
  }
}

// ---- main fused kernel -----------------------------------------------------
__global__ __launch_bounds__(NWAVES * 32)
void nerf_mlp_wmma(const float* __restrict__ x, const float* __restrict__ views,
                   const float* __restrict__ keep, const _Float16* __restrict__ pw,
                   float* __restrict__ out, int nrows) {
  __shared__ __align__(16) _Float16 sW[PW_TOTAL];
  __shared__ __align__(16) _Float16 sStage[NWAVES][2][16 * 64];
  __shared__ __align__(16) float    sF[NWAVES][16 * 32];

  const int tid = threadIdx.x, lane = tid & 31, wave = tid >> 5;

  // one-time cooperative copy of packed f16 weights into LDS
  {
    const uint4* src = (const uint4*)pw;
    uint4* dst = (uint4*)sW;
    for (int i = tid; i < PW_TOTAL / 8; i += blockDim.x) dst[i] = src[i];
  }
  __syncthreads();

  _Float16* st0 = sStage[wave][0];
  _Float16* st1 = sStage[wave][1];
  float* sf = sF[wave];

  for (int t = 0; t < TPW; ++t) {
    const int tile = blockIdx.x * (NWAVES * TPW) + wave * TPW + t;
    const int row0 = tile * 16;
    if (row0 >= nrows) break;

    // ---- prefetch next tile's inputs (overlaps with this tile's WMMAs) ----
    if (t + 1 < TPW) {
      const int nrow0 = row0 + 16;
      if (nrow0 < nrows) {
        // 32 lanes x 64B = the entire next 2KB xyz tile
        __builtin_prefetch(x + (size_t)nrow0 * 32 + lane * 16, 0, 0);
        if (lane == 0) {
          __builtin_prefetch(views + (size_t)nrow0 * 3, 0, 0);
          __builtin_prefetch(keep + nrow0, 0, 0);
        }
      }
    }

    // ---- hoisted small loads: views + keep_mask into registers early ----
    float vw0 = 0.0f, vw1 = 0.0f, vw2 = 0.0f, km = 0.0f;
    if (lane < 16) {
      const float* vp = views + (size_t)(row0 + lane) * 3;
      vw0 = vp[0]; vw1 = vp[1]; vw2 = vp[2];
      km = keep[row0 + lane];
    }

    // ---- stage embedded_xyz tile [16][32] f32 -> f16, fully coalesced ----
    {
      const int r = lane >> 1, c0 = (lane & 1) * 16;
      const float4* xp = (const float4*)(x + (size_t)(row0 + r) * 32 + c0);
      float4 f0 = xp[0], f1 = xp[1], f2 = xp[2], f3 = xp[3];
      v16h h;
      h[0]=(_Float16)f0.x; h[1]=(_Float16)f0.y; h[2]=(_Float16)f0.z; h[3]=(_Float16)f0.w;
      h[4]=(_Float16)f1.x; h[5]=(_Float16)f1.y; h[6]=(_Float16)f1.z; h[7]=(_Float16)f1.w;
      h[8]=(_Float16)f2.x; h[9]=(_Float16)f2.y; h[10]=(_Float16)f2.z; h[11]=(_Float16)f2.w;
      h[12]=(_Float16)f3.x; h[13]=(_Float16)f3.y; h[14]=(_Float16)f3.z; h[15]=(_Float16)f3.w;
      *(v16h*)(st0 + r * 32 + c0) = h;
    }
    lds_fence();

    layer_h64<1, true>(sW + OFF_S0, st0, 32, st1, lane);   // 32 -> 64 relu
    lds_fence();
    layer_h64<2, true>(sW + OFF_S1, st1, 64, st0, lane);   // 64 -> 64 relu
    lds_fence();
    layer_s2(sW + OFF_S2, st0, sf, lane);                  // 64 -> 17 (f32)
    lds_fence();

    // raw sigma (pre-softplus) per row, kept in registers by lanes 0-15
    float sigma_raw = (lane < 16) ? sf[lane * 32 + 0] : 0.0f;

    // ---- build hc = [views(3) | geo(15) | 0-pad] as f16 [16][32] ----
    if (lane < 16) {
      const int row = lane;
      _Float16* hp = st1 + row * 32;
      hp[0] = (_Float16)vw0;
      hp[1] = (_Float16)vw1;
      hp[2] = (_Float16)vw2;
#pragma unroll
      for (int j = 3; j < 18; ++j) hp[j] = (_Float16)sf[row * 32 + (j - 1)];
#pragma unroll
      for (int j = 18; j < 32; ++j) hp[j] = (_Float16)0.0f;
    }
    lds_fence();

    layer_h64<1, true>(sW + OFF_C0, st1, 32, st0, lane);   // 18(pad32) -> 64 relu
    lds_fence();
    layer_h64<2, true>(sW + OFF_C1, st0, 64, st1, lane);   // 64 -> 64 relu
    lds_fence();
    layer_h64<2, true>(sW + OFF_C2, st1, 64, st0, lane);   // 64 -> 64 relu
    lds_fence();

    // ---- c3: 64 -> 3 (single n-tile, keep cols 0..2), raw logits to LDS ----
    {
      v16h a0 = loadA(st0, 64, 0, lane), a1 = loadA(st0, 64, 1, lane);
      v8f acc = {};
      acc = WMMA_F16(a0, loadB(sW + OFF_C3, 0, lane), acc);
      acc = WMMA_F16(a1, loadB(sW + OFF_C3, 1, lane), acc);
      const int col = lane & 15;
      const int rbase = (lane >> 4) << 3;
      if (col < 3) {
#pragma unroll
        for (int r = 0; r < 8; ++r) sf[(rbase + r) * 4 + col] = acc[r];
      }
    }
    lds_fence();

    // ---- epilogue: softplus / sigmoid / mask / normalize, float4 store ----
    if (lane < 16) {
      const int row = lane;
      float xs = sigma_raw;
      float sp = (xs > 20.0f) ? xs : log1pf(__expf(xs));
      float bgs = sp * km;
      float st = bgs + 1e-9f;
      float scale = (bgs / st) * km;   // (bg_sigma/sigma_tot) * keep applied to color
      float4 o;
      o.x = scale / (1.0f + __expf(-sf[row * 4 + 0]));
      o.y = scale / (1.0f + __expf(-sf[row * 4 + 1]));
      o.z = scale / (1.0f + __expf(-sf[row * 4 + 2]));
      o.w = st;
      *(float4*)(out + (size_t)(row0 + row) * 4) = o;
    }
    lds_fence();
  }
}

// ---------------------------------------------------------------------------
extern "C" void kernel_launch(void* const* d_in, const int* in_sizes, int n_in,
                              void* d_out, int out_size, void* d_ws, size_t ws_size,
                              hipStream_t stream) {
  const float* x     = (const float*)d_in[0];   // (N,32)
  const float* views = (const float*)d_in[1];   // (N,3)
  const float* keep  = (const float*)d_in[2];   // (N,)
  const float* w_s0  = (const float*)d_in[3];   // (32,64)
  const float* w_s1  = (const float*)d_in[4];   // (64,64)
  const float* w_s2  = (const float*)d_in[5];   // (64,17)
  const float* w_c0  = (const float*)d_in[6];   // (18,64)
  const float* w_c1  = (const float*)d_in[7];   // (64,64)
  const float* w_c2  = (const float*)d_in[8];   // (64,64)
  const float* w_c3  = (const float*)d_in[9];   // (64,3)

  _Float16* pw = (_Float16*)d_ws;

  pack_weights<<<8,  256, 0, stream>>>(w_s0, pw + OFF_S0, 32, 64, 1, 4);
  pack_weights<<<16, 256, 0, stream>>>(w_s1, pw + OFF_S1, 64, 64, 2, 4);
  pack_weights<<<8,  256, 0, stream>>>(w_s2, pw + OFF_S2, 64, 17, 2, 2);
  pack_weights<<<8,  256, 0, stream>>>(w_c0, pw + OFF_C0, 18, 64, 1, 4);
  pack_weights<<<16, 256, 0, stream>>>(w_c1, pw + OFF_C1, 64, 64, 2, 4);
  pack_weights<<<16, 256, 0, stream>>>(w_c2, pw + OFF_C2, 64, 64, 2, 4);
  pack_weights<<<4,  256, 0, stream>>>(w_c3, pw + OFF_C3, 64, 3, 2, 1);

  const int nrows = in_sizes[0] / 32;                 // 2,097,152
  const int rows_per_block = NWAVES * TPW * 16;       // 512
  const int grid = (nrows + rows_per_block - 1) / rows_per_block;

  nerf_mlp_wmma<<<grid, NWAVES * 32, 0, stream>>>(x, views, keep, pw,
                                                  (float*)d_out, nrows);
}
